// QGRU_25838523253263
// MI455X (gfx1250) — matrix operand order repaired and checked by reading, not compiled, not run on hardware
//
#include <hip/hip_runtime.h>
#include <hip/hip_bf16.h>
#include <math.h>

// CDNA5 / gfx1250 WMMA types (wave32)
typedef __bf16 v16bf __attribute__((ext_vector_type(16)));
typedef __bf16 v8bf  __attribute__((ext_vector_type(8)));
typedef float  v8f   __attribute__((ext_vector_type(8)));

#define IN_R   64     // 4*IN_Q  (real input dim)
#define HID    256    // 4*H_Q   (real hidden dim)
#define BATCH  128
#define TLEN   2048
#define BM     16     // batch rows per workgroup (== WMMA M)
#define NWAVES 16
#define NTHREADS (NWAVES * 32)

// LDS carve-up (dynamic LDS: 320KB/WGP on CDNA5, static __shared__ caps at 64KB)
#define HF_STRIDE   (HID + 4)                 // padded f32 h row
#define HBF_STRIDE  (HID + 8)                 // padded bf16 h row
#define XSB_STRIDE  (IN_R + 8)                // padded bf16 x row
#define HF_BYTES    (BM * HF_STRIDE * 4)                    // 16640
#define WHL_ELEMS   (2 * NWAVES * 8 * 32 * 16)              // gates f,i fragments
#define WHL_BYTES   (WHL_ELEMS * 2)                         // 262144
#define HBF_BYTES   (BM * HBF_STRIDE * 2)                   // 8448
#define XSB_BYTES   (2 * BM * XSB_STRIDE * 2)               // 4608
#define SMEM_BYTES  (HF_BYTES + WHL_BYTES + HBF_BYTES + XSB_BYTES)  // 291840 < 320K

// Element of the big quaternion block matrix:
//   W_big[inq*a + p, 256*g + 64*b + q] = sign(a,b) * W_{a^b}[p, g*64 + q]
// sign mask: bit (a*4+b) of 0x5390 set -> negative.
__device__ __forceinline__ float qbig_elem(const float* Wr, const float* Wi,
                                           const float* Wj, const float* Wk,
                                           int inq, int row, int col) {
    const int a  = row / inq;
    const int p  = row - a * inq;
    const int g  = col >> 8;          // gate 0..2
    const int cg = col & 255;
    const int b  = cg >> 6;
    const int q  = cg & 63;
    const int cp = a ^ b;             // component select: r,i,j,k
    const float* W = (cp == 0) ? Wr : (cp == 1) ? Wi : (cp == 2) ? Wj : Wk;
    const float v = W[p * 192 + g * 64 + q];   // each W_* is (inq, 192) row-major
    return ((0x5390u >> (a * 4 + b)) & 1u) ? -v : v;
}

__device__ __forceinline__ float sigmoidf_(float v) {
    return 1.f / (1.f + __expf(-v));
}

__global__ __launch_bounds__(NTHREADS, 1)
void qgru_persistent(const float* __restrict__ x,
                     const float* __restrict__ Wx_r, const float* __restrict__ Wx_i,
                     const float* __restrict__ Wx_j, const float* __restrict__ Wx_k,
                     const float* __restrict__ bx,
                     const float* __restrict__ Wh_r, const float* __restrict__ Wh_i,
                     const float* __restrict__ Wh_j, const float* __restrict__ Wh_k,
                     const float* __restrict__ bh,
                     float* __restrict__ out)
{
    extern __shared__ __attribute__((aligned(64))) unsigned char smem_raw[];
    float  (*h_f)[HF_STRIDE]   = (float  (*)[HF_STRIDE])(smem_raw);
    __bf16 (*WhL)[NWAVES][8][32][16] =
        (__bf16 (*)[NWAVES][8][32][16])(smem_raw + HF_BYTES);
    __bf16 (*h_bf)[HBF_STRIDE] = (__bf16 (*)[HBF_STRIDE])(smem_raw + HF_BYTES + WHL_BYTES);
    __bf16 (*xsb)[BM][XSB_STRIDE] =
        (__bf16 (*)[BM][XSB_STRIDE])(smem_raw + HF_BYTES + WHL_BYTES + HBF_BYTES);

    const int tid   = threadIdx.x;
    const int w     = tid >> 5;        // wave id: owns hidden cols 16w..16w+15
    const int lane  = tid & 31;
    const int n     = lane & 15;       // N index within tile / A row
    const int half  = lane >> 4;
    const int koffA = half * 16;       // A-fragment K offset pattern (contiguous: K = 16*half + e)
    const int bm    = blockIdx.x * BM; // batch row base

    // ---- gates f,i of Wh_big -> LDS in fragment-linear layout ----
    // B layout: lane<16 holds K 0..15, lane>=16 holds K 16..31; N = lane&15.
    for (int i = tid; i < WHL_ELEMS; i += NTHREADS) {
        const int e  = i & 15;
        const int ln = (i >> 4) & 31;
        const int ks = (i >> 9) & 7;
        const int wv = (i >> 12) & 15;
        const int g  = i >> 16;
        const int k  = ks * 32 + (ln >> 4) * 16 + e;
        WhL[g][wv][ks][ln][e] = (__bf16)qbig_elem(Wh_r, Wh_i, Wh_j, Wh_k,
                                                  64, k, g * 256 + wv * 16 + (ln & 15));
    }

    // ---- gate n of Wh_big and all of Wx_big -> persistent VGPR fragments ----
    v16bf Bhn[8];     // gate n, K=256 -> 8 k-steps (64 VGPRs)
    v16bf Bx[3][2];   // 3 gates x (K=64 -> 2 k-steps) (48 VGPRs)
#pragma unroll
    for (int ks = 0; ks < 8; ++ks) {
#pragma unroll
        for (int e = 0; e < 16; ++e) {
            const int k = ks * 32 + half * 16 + e;
            Bhn[ks][e] = (__bf16)qbig_elem(Wh_r, Wh_i, Wh_j, Wh_k,
                                           64, k, 2 * 256 + w * 16 + n);
        }
    }
#pragma unroll
    for (int g = 0; g < 3; ++g) {
#pragma unroll
        for (int ks = 0; ks < 2; ++ks) {
#pragma unroll
            for (int e = 0; e < 16; ++e) {
                const int k = ks * 32 + half * 16 + e;
                Bx[g][ks][e] = (__bf16)qbig_elem(Wx_r, Wx_i, Wx_j, Wx_k,
                                                 16, k, g * 256 + w * 16 + n);
            }
        }
    }

    const int   c      = w * 16 + n;        // this lane's hidden column
    const float biasF  = bx[c]       + bh[c];
    const float biasI  = bx[256 + c] + bh[256 + c];
    const float biasXN = bx[512 + c];
    const float biasHN = bh[512 + c];

    // ---- h0 = 0 ----
    for (int i = tid; i < BM * HID; i += NTHREADS) {
        const int r = i >> 8, cc = i & 255;
        h_f[r][cc]  = 0.f;
        h_bf[r][cc] = (__bf16)0.f;
    }
    // ---- stage x[t=0] (converted to bf16) ----
    {
        const int idx = tid * 2, r = idx >> 6, cc = idx & 63;
        const float2 v = *(const float2*)&x[((size_t)(bm + r) * TLEN + 0) * IN_R + cc];
        xsb[0][r][cc] = (__bf16)v.x; xsb[0][r][cc + 1] = (__bf16)v.y;
    }
    __syncthreads();

    for (int t = 0; t < TLEN; ++t) {
        const int cur = t & 1, nxt = cur ^ 1;

        // prefetch x[t+1] into registers; latency hides under 30 WMMAs
        float2 pf = make_float2(0.f, 0.f);
        const int pidx = tid * 2, pr = pidx >> 6, pc = pidx & 63;
        if (t + 1 < TLEN)
            pf = *(const float2*)&x[((size_t)(bm + pr) * TLEN + (t + 1)) * IN_R + pc];
        if (t + 2 < TLEN)
            __builtin_prefetch(&x[((size_t)(bm + pr) * TLEN + (t + 2)) * IN_R + pc], 0, 1);

        v8f accF = {}, accI = {}, accXN = {}, accHN = {};

        // gx = x_t @ Wx_big  (K = 64 -> 2 WMMA k-steps, 3 gates)
#pragma unroll
        for (int ks = 0; ks < 2; ++ks) {
            v16bf Ax;
            const v8bf xlo = *(const v8bf*)&xsb[cur][n][ks * 32 + koffA];
            const v8bf xhi = *(const v8bf*)&xsb[cur][n][ks * 32 + 8 + koffA];
#pragma unroll
            for (int e = 0; e < 8; ++e) { Ax[e] = xlo[e]; Ax[8 + e] = xhi[e]; }
            accF  = __builtin_amdgcn_wmma_f32_16x16x32_bf16(false, Ax, false, Bx[0][ks], (short)0, accF,  false, false);
            accI  = __builtin_amdgcn_wmma_f32_16x16x32_bf16(false, Ax, false, Bx[1][ks], (short)0, accI,  false, false);
            accXN = __builtin_amdgcn_wmma_f32_16x16x32_bf16(false, Ax, false, Bx[2][ks], (short)0, accXN, false, false);
        }

        // gh = h @ Wh_big  (K = 256 -> 8 WMMA k-steps, 3 gates)
        // gates f,i stream their B fragments from LDS (2x ds_load_b128 each,
        // conflict-free fragment-linear layout); gate n uses register fragments.
#pragma unroll
        for (int ks = 0; ks < 8; ++ks) {
            v16bf Ah;
            const v8bf lo = *(const v8bf*)&h_bf[n][ks * 32 + koffA];
            const v8bf hi = *(const v8bf*)&h_bf[n][ks * 32 + 8 + koffA];
#pragma unroll
            for (int e = 0; e < 8; ++e) { Ah[e] = lo[e]; Ah[8 + e] = hi[e]; }
            const v16bf Bf = *(const v16bf*)&WhL[0][w][ks][lane][0];
            const v16bf Bi = *(const v16bf*)&WhL[1][w][ks][lane][0];
            accF  = __builtin_amdgcn_wmma_f32_16x16x32_bf16(false, Ah, false, Bf,      (short)0, accF,  false, false);
            accI  = __builtin_amdgcn_wmma_f32_16x16x32_bf16(false, Ah, false, Bi,      (short)0, accI,  false, false);
            accHN = __builtin_amdgcn_wmma_f32_16x16x32_bf16(false, Ah, false, Bhn[ks], (short)0, accHN, false, false);
        }

        __syncthreads();   // all waves finished reading h_bf / xsb[cur]

        // gates + state update for this wave's 16(batch) x 16(hidden) tile.
        // C/D layout: VGPR r, lanes 0-15 -> M=r, lanes 16-31 -> M=r+8, N=lane&15.
#pragma unroll
        for (int r = 0; r < 8; ++r) {
            const int   m  = r + half * 8;
            const float ho = h_f[m][c];
            const float rt = sigmoidf_(accF[r] + biasF);
            const float zt = sigmoidf_(accI[r] + biasI);
            float nt = accXN[r] + biasXN + rt * (accHN[r] + biasHN);
            nt = fmaxf(nt, 0.f);
            const float hn = zt * ho + (1.f - zt) * nt;
            h_f[m][c]  = hn;
            h_bf[m][c] = (__bf16)hn;
        }

        if (t + 1 < TLEN) {
            xsb[nxt][pr][pc]     = (__bf16)pf.x;
            xsb[nxt][pr][pc + 1] = (__bf16)pf.y;
        }

        __syncthreads();   // h_new visible before next step's A-fragment reads
    }

    // write hT (B, 256) f32
    for (int i = tid; i < BM * HID; i += NTHREADS) {
        const int r = i >> 8, cc = i & 255;
        out[(size_t)(bm + r) * HID + cc] = h_f[r][cc];
    }
}

extern "C" void kernel_launch(void* const* d_in, const int* in_sizes, int n_in,
                              void* d_out, int out_size, void* d_ws, size_t ws_size,
                              hipStream_t stream) {
    (void)in_sizes; (void)n_in; (void)out_size; (void)d_ws; (void)ws_size;
    const float* xin  = (const float*)d_in[0];
    const float* Wx_r = (const float*)d_in[1];
    const float* Wx_i = (const float*)d_in[2];
    const float* Wx_j = (const float*)d_in[3];
    const float* Wx_k = (const float*)d_in[4];
    const float* bx   = (const float*)d_in[5];
    const float* Wh_r = (const float*)d_in[6];
    const float* Wh_i = (const float*)d_in[7];
    const float* Wh_j = (const float*)d_in[8];
    const float* Wh_k = (const float*)d_in[9];
    const float* bh   = (const float*)d_in[10];
    float* out = (float*)d_out;

    // Allow the large dynamic-LDS allocation (capture-safe: not a stream op).
    (void)hipFuncSetAttribute((const void*)qgru_persistent,
                              hipFuncAttributeMaxDynamicSharedMemorySize,
                              SMEM_BYTES);

    qgru_persistent<<<BATCH / BM, NTHREADS, SMEM_BYTES, stream>>>(
        xin, Wx_r, Wx_i, Wx_j, Wx_k, bx, Wh_r, Wh_i, Wh_j, Wh_k, bh, out);
}
